// MultiViewEntropyBottleneck_77807627534750
// MI455X (gfx1250) — matrix operand order repaired and checked by reading, not compiled, not run on hardware
//
#include <hip/hip_runtime.h>
#include <hip/hip_bf16.h>
#include <math.h>

// ---------------------------------------------------------------------------
// CDNA5 (gfx1250) wave32 WMMA types
// ---------------------------------------------------------------------------
typedef __attribute__((ext_vector_type(16))) __bf16 v16bf;
typedef __attribute__((ext_vector_type(8)))  __bf16 v8bf;
typedef __attribute__((ext_vector_type(8)))  float  v8f;

static __device__ __forceinline__ v8f wmma_bf16(v16bf a, v16bf b, v8f c) {
  // D = A(16x32 bf16) * B(32x16 bf16) + C(16x16 f32)
  return __builtin_amdgcn_wmma_f32_16x16x32_bf16(
      /*neg_a=*/false, a, /*neg_b=*/false, b,
      /*c_mod=*/(short)0, c, /*reuse_a=*/false, /*reuse_b=*/false);
}

static __device__ __forceinline__ v16bf cat8(v8bf lo, v8bf hi) {
  return __builtin_shufflevector(lo, hi, 0, 1, 2, 3, 4, 5, 6, 7,
                                         8, 9, 10, 11, 12, 13, 14, 15);
}

static __device__ __forceinline__ float block_reduce_sum_256(float v, float* red) {
  const int t = threadIdx.x;
  red[t] = v;
  __syncthreads();
  #pragma unroll
  for (int s = 128; s > 0; s >>= 1) {
    if (t < s) red[t] += red[t + s];
    __syncthreads();
  }
  float r = red[0];
  __syncthreads();
  return r;
}

// ---------------------------------------------------------------------------
// Weight transpose + convert: Wt[n*K + k] = bf16(W[k*N + n])  (col-major bf16)
// grid = (N/32, K/32), block = 256 (32x8), LDS tiled for coalescing.
// ---------------------------------------------------------------------------
__global__ __launch_bounds__(256) void wt_transpose_kernel(
    const float* __restrict__ W, __bf16* __restrict__ Wt, int K, int N)
{
  __shared__ float tile[32][33];
  const int tx = threadIdx.x & 31;
  const int ty = threadIdx.x >> 5;           // 0..7
  const int n0 = blockIdx.x * 32;
  const int k0 = blockIdx.y * 32;

  #pragma unroll
  for (int r = 0; r < 32; r += 8)
    tile[r + ty][tx] = W[(size_t)(k0 + r + ty) * N + (n0 + tx)];
  __syncthreads();
  #pragma unroll
  for (int r = 0; r < 32; r += 8)
    Wt[(size_t)(n0 + r + ty) * K + (k0 + tx)] = (__bf16)tile[tx][r + ty];
}

// ---------------------------------------------------------------------------
// Gather last-token rows (seq_last = sum(mask) - 1); features -> bf16.
// grid = 256 (one block per batch row), block = 256
// ---------------------------------------------------------------------------
__global__ __launch_bounds__(256) void gather_last_kernel(
    const float* __restrict__ img, const float* __restrict__ txt,
    const int* __restrict__ mask,
    const float* __restrict__ epsi, const float* __restrict__ epst,
    __bf16* __restrict__ Xi, __bf16* __restrict__ Xt,
    float* __restrict__ Ei, float* __restrict__ Et)
{
  const int b = blockIdx.x;
  const int t = threadIdx.x;
  __shared__ int red[128];
  __shared__ int lsh;
  if (t < 128) red[t] = mask[b * 128 + t];
  __syncthreads();
  #pragma unroll
  for (int s = 64; s > 0; s >>= 1) {
    if (t < s) red[t] += red[t + s];
    __syncthreads();
  }
  if (t == 0) lsh = red[0] - 1;
  __syncthreads();
  const size_t tok = (size_t)b * 128 + lsh;

  for (int i = t; i < 768; i += 256) {
    Xi[(size_t)b * 768 + i] = (__bf16)img[tok * 768 + i];
    Xt[(size_t)b * 768 + i] = (__bf16)txt[tok * 768 + i];
  }
  Ei[(size_t)b * 256 + t] = epsi[tok * 256 + t];
  Et[(size_t)b * 256 + t] = epst[tok * 256 + t];
}

// ---------------------------------------------------------------------------
// WMMA GEMM: out(MxN) = A(MxK,bf16 row-major) @ Wt(NxK,bf16 col-major) + bias.
// fp32 output, optional bf16 mirror.  grid.x = M/16; block = 256 (8 waves),
// each wave owns TPW n-tiles.  Requires N == TPW*8*16, K % 32 == 0.
// Inner loop is pure vector-load -> v_wmma (no conversion VALU).
// ---------------------------------------------------------------------------
template <int TPW>
__global__ __launch_bounds__(256) void wmma_gemm_bias(
    const __bf16* __restrict__ A, const __bf16* __restrict__ Wt,
    const float* __restrict__ bias, float* __restrict__ out,
    __bf16* __restrict__ out_bf, int K, int N)
{
  const int tid   = threadIdx.x;
  const int wave  = tid >> 5;
  const int lane  = tid & 31;
  const int lrow  = lane & 15;   // N index within tile (B/C), M row for A
  const int lhalf = lane >> 4;   // K-half selector
  const int m0    = blockIdx.x * 16;

  v8f acc[TPW];
  #pragma unroll
  for (int j = 0; j < TPW; ++j) acc[j] = (v8f){0.f,0.f,0.f,0.f,0.f,0.f,0.f,0.f};

  for (int k0 = 0; k0 < K; k0 += 32) {
    // A fragment: 16 contiguous K values per lane (matches B), two aligned 16B chunks
    const __bf16* arow = A + (size_t)(m0 + lrow) * K + k0 + lhalf * 16;
    __builtin_prefetch(arow + 32, 0, 0);     // global_prefetch_b8 next k-block
    const v16bf afrag = cat8(*(const v8bf*)arow, *(const v8bf*)(arow + 8));

    #pragma unroll
    for (int j = 0; j < TPW; ++j) {
      const int n = (wave + 8 * j) * 16 + lrow;
      // B fragment: 16 contiguous K values (col-major weight), one 32B load
      const v16bf bfrag = *(const v16bf*)(Wt + (size_t)n * K + k0 + lhalf * 16);
      acc[j] = wmma_bf16(afrag, bfrag, acc[j]);
    }
  }

  #pragma unroll
  for (int j = 0; j < TPW; ++j) {
    const int n = (wave + 8 * j) * 16 + lrow;
    const float bn = bias[n];
    #pragma unroll
    for (int v = 0; v < 8; ++v) {
      const int m = m0 + lhalf * 8 + v;
      const float val = acc[j][v] + bn;
      out[(size_t)m * N + n] = val;
      if (out_bf) out_bf[(size_t)m * N + n] = (__bf16)val;
    }
  }
}

// ---------------------------------------------------------------------------
// Row-wise LayerNorm + exact GELU: reads fp32 GEMM output, writes bf16
// activations for the next WMMA GEMM.  grid = M rows, block = 256.
// ---------------------------------------------------------------------------
__global__ __launch_bounds__(256) void ln_gelu_kernel(
    const float* __restrict__ buf, __bf16* __restrict__ outb,
    const float* __restrict__ g, const float* __restrict__ beta, int N)
{
  __shared__ float red[256];
  const int t = threadIdx.x;
  const float* row = buf + (size_t)blockIdx.x * N;
  __bf16* orow = outb + (size_t)blockIdx.x * N;

  float s = 0.f;
  for (int i = t; i < N; i += 256) s += row[i];
  const float mean = block_reduce_sum_256(s, red) / (float)N;

  float s2 = 0.f;
  for (int i = t; i < N; i += 256) { float d = row[i] - mean; s2 += d * d; }
  const float var  = block_reduce_sum_256(s2, red) / (float)N;
  const float rstd = rsqrtf(var + 1e-5f);

  for (int i = t; i < N; i += 256) {
    float h = (row[i] - mean) * rstd * g[i] + beta[i];
    orow[i] = (__bf16)(0.5f * h * (1.0f + erff(h * 0.70710678118654752f)));
  }
}

// ---------------------------------------------------------------------------
// Reparameterize + L2 normalize + per-row KL.  grid = 256 rows, block = 256.
// P row = [mu(256) | logvar(256)].  Writes fp32 z (row stats) + bf16 z (WMMA).
// ---------------------------------------------------------------------------
__global__ __launch_bounds__(256) void reparam_kernel(
    const float* __restrict__ P, const float* __restrict__ E,
    float* __restrict__ Z, __bf16* __restrict__ Zb, float* __restrict__ KL)
{
  __shared__ float red[256];
  const int r = blockIdx.x, t = threadIdx.x;
  const float mu = P[(size_t)r * 512 + t];
  const float lv = P[(size_t)r * 512 + 256 + t];
  const float v  = mu + E[(size_t)r * 256 + t] * expf(0.5f * lv);

  const float ss  = block_reduce_sum_256(v * v, red);
  const float nrm = fmaxf(sqrtf(ss), 1e-12f);
  const float z   = v / nrm;
  Z [(size_t)r * 256 + t] = z;
  Zb[(size_t)r * 256 + t] = (__bf16)z;

  const float kle = 1.0f + lv - mu * mu - expf(lv);
  const float kls = block_reduce_sum_256(kle, red);
  if (t == 0) KL[r] = -0.5f * kls;
}

// ---------------------------------------------------------------------------
// Row stats: x2 = |x|^2, y2 = |y|^2, ysy = y·sy.  grid = 256, block = 256.
// ---------------------------------------------------------------------------
__global__ __launch_bounds__(256) void rowstats_kernel(
    const float* __restrict__ X, const float* __restrict__ Y,
    const float* __restrict__ SY,
    float* __restrict__ x2, float* __restrict__ y2, float* __restrict__ ysy)
{
  __shared__ float red[256];
  const int r = blockIdx.x, t = threadIdx.x;
  const float xv = X[(size_t)r * 256 + t];
  const float yv = Y[(size_t)r * 256 + t];
  const float sv = SY[(size_t)r * 256 + t];
  float a = block_reduce_sum_256(xv * xv, red);
  float b = block_reduce_sum_256(yv * yv, red);
  float c = block_reduce_sum_256(yv * sv, red);
  if (t == 0) { x2[r] = a; y2[r] = b; ysy[r] = c; }
}

// ---------------------------------------------------------------------------
// Pairwise kernel: three 256x256x256 A@B^T WMMA GEMMs (all-bf16 inputs):
//   Gxy = X @ Y^T, Gss = SX @ SY^T, Gxsy = X @ SY^T
// emits dist^2 = max(x2+y2-2*Gxy, 0) plus Gss, Gxsy.
// grid = 16 (16 rows each), block = 256 (8 waves x 2 n-tiles).
// ---------------------------------------------------------------------------
__global__ __launch_bounds__(256) void pairwise_kernel(
    const __bf16* __restrict__ X,  const __bf16* __restrict__ SX,
    const __bf16* __restrict__ Y,  const __bf16* __restrict__ SY,
    const float* __restrict__ x2, const float* __restrict__ y2,
    float* __restrict__ Dist, float* __restrict__ Gss, float* __restrict__ Gxsy)
{
  const int tid = threadIdx.x, wave = tid >> 5, lane = tid & 31;
  const int lrow = lane & 15, lhalf = lane >> 4;
  const int m0 = blockIdx.x * 16;
  const int D = 256;

  const v8f z8 = (v8f){0.f,0.f,0.f,0.f,0.f,0.f,0.f,0.f};
  v8f axy[2] = {z8, z8}, ass[2] = {z8, z8}, axs[2] = {z8, z8};

  for (int k0 = 0; k0 < D; k0 += 32) {
    const __bf16* xr = X  + (size_t)(m0 + lrow) * D + k0 + lhalf * 16;
    const __bf16* sr = SX + (size_t)(m0 + lrow) * D + k0 + lhalf * 16;
    const v16bf az = cat8(*(const v8bf*)xr, *(const v8bf*)(xr + 8));
    const v16bf as = cat8(*(const v8bf*)sr, *(const v8bf*)(sr + 8));

    #pragma unroll
    for (int j = 0; j < 2; ++j) {
      const int n = (wave + 8 * j) * 16 + lrow;
      // B^T: contiguous 32B per lane
      const v16bf bz = *(const v16bf*)(Y  + (size_t)n * D + k0 + lhalf * 16);
      const v16bf bs = *(const v16bf*)(SY + (size_t)n * D + k0 + lhalf * 16);
      axy[j] = wmma_bf16(az, bz, axy[j]);
      ass[j] = wmma_bf16(as, bs, ass[j]);
      axs[j] = wmma_bf16(az, bs, axs[j]);
    }
  }

  #pragma unroll
  for (int j = 0; j < 2; ++j) {
    const int n = (wave + 8 * j) * 16 + lrow;
    const float yn = y2[n];
    #pragma unroll
    for (int v = 0; v < 8; ++v) {
      const int m = m0 + lhalf * 8 + v;
      const size_t idx = (size_t)m * 256 + n;
      const float d = fmaxf(x2[m] + yn - 2.0f * axy[j][v], 0.0f);
      Dist[idx] = d;
      Gss[idx]  = ass[j][v];
      Gxsy[idx] = axs[j][v];
    }
  }
}

// ---------------------------------------------------------------------------
// Exact median of 65536 non-negative floats via 32-bit radix select on IEEE
// bits (order-preserving for >= 0).  jnp.median = avg of ranks 32767/32768.
// One block, 256 threads.  Writes bw^2 (clipped bandwidth squared).
// ---------------------------------------------------------------------------
__global__ __launch_bounds__(256) void median_bw_kernel(
    const float* __restrict__ dist, float* __restrict__ bw2out)
{
  __shared__ unsigned red[256];
  const int tid = threadIdx.x;
  float mv[2];
  const int ks[2] = {32767, 32768};

  for (int sel = 0; sel < 2; ++sel) {
    unsigned prefix = 0;
    int k = ks[sel];
    for (int bit = 31; bit >= 0; --bit) {
      const unsigned hmask = (bit == 31) ? 0u : (0xFFFFFFFFu << (bit + 1));
      unsigned cnt = 0;
      for (int i = tid; i < 65536; i += 256) {
        const unsigned u = __float_as_uint(dist[i]);
        if ((u & hmask) == prefix && ((u >> bit) & 1u) == 0u) cnt++;
      }
      red[tid] = cnt;
      __syncthreads();
      #pragma unroll
      for (int s = 128; s > 0; s >>= 1) {
        if (tid < s) red[tid] += red[tid + s];
        __syncthreads();
      }
      const int c0 = (int)red[0];
      __syncthreads();
      if (k >= c0) { k -= c0; prefix |= (1u << bit); }
    }
    mv[sel] = __uint_as_float(prefix);
  }

  if (tid == 0) {
    const float med = 0.5f * (mv[0] + mv[1]);
    float bw = sqrtf(med * 0.5f) * 1.0f;            // BW_FACTOR
    bw = fminf(fmaxf(bw, 0.1f), 10.0f);             // clip
    bw2out[0] = bw * bw;
  }
}

// ---------------------------------------------------------------------------
// Final reduction: alignment mean + KL means -> total loss scalar.
// One block, 256 threads.
// ---------------------------------------------------------------------------
__global__ __launch_bounds__(256) void finalize_kernel(
    const float* __restrict__ Dist, const float* __restrict__ Gss,
    const float* __restrict__ Gxsy, const float* __restrict__ ysy,
    const float* __restrict__ KLi,  const float* __restrict__ KLt,
    const float* __restrict__ bw2p, float* __restrict__ out)
{
  __shared__ float red[256];
  const int tid = threadIdx.x;
  const float bw2  = bw2p[0];
  const float inv  = 1.0f / bw2;
  const float inv2 = inv * inv;
  const float Df   = 256.0f;

  float s = 0.f;
  for (int i = tid; i < 65536; i += 256) {
    const int n = i & 255;
    const float d  = Dist[i];
    const float kk = expf(-d * (0.5f * inv));
    const float t1 = kk * Gss[i];
    const float t2 = (Gxsy[i] - ysy[n]) * kk * inv;
    const float t3 = d * kk * inv2 - kk * Df * inv;
    s += t1 + t2 + t3;
  }
  const float align_sum = block_reduce_sum_256(s, red);
  const float kl_sum    = block_reduce_sum_256(KLi[tid] + KLt[tid], red);

  if (tid == 0) {
    const float kl_loss    = kl_sum / 512.0f;            // (mean+mean)/2
    const float align_loss = align_sum / 65536.0f;
    out[0] = kl_loss + 0.1f * align_loss;                // BETA = 0.1
  }
}

// ---------------------------------------------------------------------------
// Host launcher
// ---------------------------------------------------------------------------
extern "C" void kernel_launch(void* const* d_in, const int* in_sizes, int n_in,
                              void* d_out, int out_size, void* d_ws, size_t ws_size,
                              hipStream_t stream) {
  (void)in_sizes; (void)n_in; (void)out_size; (void)ws_size;

  const float* img  = (const float*)d_in[0];
  const float* txt  = (const float*)d_in[1];
  const float* epsi = (const float*)d_in[2];
  const float* epst = (const float*)d_in[3];
  const int*   mask = (const int*)  d_in[4];
  const float* eiW1 = (const float*)d_in[5];
  const float* eib1 = (const float*)d_in[6];
  const float* eig  = (const float*)d_in[7];
  const float* eibt = (const float*)d_in[8];
  const float* eiW2 = (const float*)d_in[9];
  const float* eib2 = (const float*)d_in[10];
  const float* etW1 = (const float*)d_in[11];
  const float* etb1 = (const float*)d_in[12];
  const float* etg  = (const float*)d_in[13];
  const float* etbt = (const float*)d_in[14];
  const float* etW2 = (const float*)d_in[15];
  const float* etb2 = (const float*)d_in[16];
  const float* siW1 = (const float*)d_in[17];
  const float* sib1 = (const float*)d_in[18];
  const float* sig  = (const float*)d_in[19];
  const float* sibt = (const float*)d_in[20];
  const float* siW2 = (const float*)d_in[21];
  const float* sib2 = (const float*)d_in[22];
  const float* stW1 = (const float*)d_in[23];
  const float* stb1 = (const float*)d_in[24];
  const float* stg  = (const float*)d_in[25];
  const float* stbt = (const float*)d_in[26];
  const float* stW2 = (const float*)d_in[27];
  const float* stb2 = (const float*)d_in[28];

  // Byte-based workspace carve-out, 256B aligned slices.
  size_t o = 0;
  char* base = (char*)d_ws;
  auto alloc = [&](size_t bytes) -> void* {
    void* p = base + o;
    o += (bytes + 255) & ~(size_t)255;
    return p;
  };
  // bf16 transposed weights (col-major N x K)
  __bf16* WtEi1 = (__bf16*)alloc(1024 * 768  * 2);
  __bf16* WtEi2 = (__bf16*)alloc(512  * 1024 * 2);
  __bf16* WtEt1 = (__bf16*)alloc(1024 * 768  * 2);
  __bf16* WtEt2 = (__bf16*)alloc(512  * 1024 * 2);
  __bf16* WtSi1 = (__bf16*)alloc(512  * 256  * 2);
  __bf16* WtSi2 = (__bf16*)alloc(256  * 512  * 2);
  __bf16* WtSt1 = (__bf16*)alloc(512  * 256  * 2);
  __bf16* WtSt2 = (__bf16*)alloc(256  * 512  * 2);
  // activations
  __bf16* Xi   = (__bf16*)alloc(256 * 768 * 2);
  __bf16* Xt   = (__bf16*)alloc(256 * 768 * 2);
  float*  Ei   = (float*) alloc(256 * 256 * 4);
  float*  Et   = (float*) alloc(256 * 256 * 4);
  float*  Hi   = (float*) alloc(256 * 1024 * 4);
  float*  Ht   = (float*) alloc(256 * 1024 * 4);
  __bf16* Hib  = (__bf16*)alloc(256 * 1024 * 2);
  __bf16* Htb  = (__bf16*)alloc(256 * 1024 * 2);
  float*  Pi   = (float*) alloc(256 * 512 * 4);
  float*  Pt   = (float*) alloc(256 * 512 * 4);
  float*  Zi   = (float*) alloc(256 * 256 * 4);
  float*  Zt   = (float*) alloc(256 * 256 * 4);
  __bf16* Zib  = (__bf16*)alloc(256 * 256 * 2);
  __bf16* Ztb  = (__bf16*)alloc(256 * 256 * 2);
  float*  S1i  = (float*) alloc(256 * 512 * 4);
  float*  S1t  = (float*) alloc(256 * 512 * 4);
  __bf16* S1ib = (__bf16*)alloc(256 * 512 * 2);
  __bf16* S1tb = (__bf16*)alloc(256 * 512 * 2);
  float*  SCi  = (float*) alloc(256 * 256 * 4);
  float*  SCt  = (float*) alloc(256 * 256 * 4);
  __bf16* SCib = (__bf16*)alloc(256 * 256 * 2);
  __bf16* SCtb = (__bf16*)alloc(256 * 256 * 2);
  float*  KLi  = (float*) alloc(256 * 4);
  float*  KLt  = (float*) alloc(256 * 4);
  float*  x2v  = (float*) alloc(256 * 4);
  float*  y2v  = (float*) alloc(256 * 4);
  float*  ysyv = (float*) alloc(256 * 4);
  float*  Dist = (float*) alloc(256 * 256 * 4);
  float*  Gss  = (float*) alloc(256 * 256 * 4);
  float*  Gxsy = (float*) alloc(256 * 256 * 4);
  float*  bw2p = (float*) alloc(64);

  // 0) Weight transpose/convert to bf16 col-major (one pass per matrix)
  wt_transpose_kernel<<<dim3(1024/32, 768/32),  256, 0, stream>>>(eiW1, WtEi1, 768, 1024);
  wt_transpose_kernel<<<dim3(512/32,  1024/32), 256, 0, stream>>>(eiW2, WtEi2, 1024, 512);
  wt_transpose_kernel<<<dim3(1024/32, 768/32),  256, 0, stream>>>(etW1, WtEt1, 768, 1024);
  wt_transpose_kernel<<<dim3(512/32,  1024/32), 256, 0, stream>>>(etW2, WtEt2, 1024, 512);
  wt_transpose_kernel<<<dim3(512/32,  256/32),  256, 0, stream>>>(siW1, WtSi1, 256, 512);
  wt_transpose_kernel<<<dim3(256/32,  512/32),  256, 0, stream>>>(siW2, WtSi2, 512, 256);
  wt_transpose_kernel<<<dim3(512/32,  256/32),  256, 0, stream>>>(stW1, WtSt1, 256, 512);
  wt_transpose_kernel<<<dim3(256/32,  512/32),  256, 0, stream>>>(stW2, WtSt2, 512, 256);

  // 1) Gather last tokens (loss only depends on seq_last positions)
  gather_last_kernel<<<256, 256, 0, stream>>>(img, txt, mask, epsi, epst, Xi, Xt, Ei, Et);

  // 2) Encoder MLPs (256x768x1024 -> LN+GELU -> 256x1024x512)
  wmma_gemm_bias<8><<<16, 256, 0, stream>>>(Xi, WtEi1, eib1, Hi, (__bf16*)nullptr, 768, 1024);
  ln_gelu_kernel<<<256, 256, 0, stream>>>(Hi, Hib, eig, eibt, 1024);
  wmma_gemm_bias<4><<<16, 256, 0, stream>>>(Hib, WtEi2, eib2, Pi, (__bf16*)nullptr, 1024, 512);

  wmma_gemm_bias<8><<<16, 256, 0, stream>>>(Xt, WtEt1, etb1, Ht, (__bf16*)nullptr, 768, 1024);
  ln_gelu_kernel<<<256, 256, 0, stream>>>(Ht, Htb, etg, etbt, 1024);
  wmma_gemm_bias<4><<<16, 256, 0, stream>>>(Htb, WtEt2, etb2, Pt, (__bf16*)nullptr, 1024, 512);

  // 3) Reparameterize + normalize + per-row KL
  reparam_kernel<<<256, 256, 0, stream>>>(Pi, Ei, Zi, Zib, KLi);
  reparam_kernel<<<256, 256, 0, stream>>>(Pt, Et, Zt, Ztb, KLt);

  // 4) Score MLPs (256x256x512 -> LN+GELU -> 256x512x256)
  wmma_gemm_bias<4><<<16, 256, 0, stream>>>(Zib, WtSi1, sib1, S1i, (__bf16*)nullptr, 256, 512);
  ln_gelu_kernel<<<256, 256, 0, stream>>>(S1i, S1ib, sig, sibt, 512);
  wmma_gemm_bias<2><<<16, 256, 0, stream>>>(S1ib, WtSi2, sib2, SCi, SCib, 512, 256);

  wmma_gemm_bias<4><<<16, 256, 0, stream>>>(Ztb, WtSt1, stb1, S1t, (__bf16*)nullptr, 256, 512);
  ln_gelu_kernel<<<256, 256, 0, stream>>>(S1t, S1tb, stg, stbt, 512);
  wmma_gemm_bias<2><<<16, 256, 0, stream>>>(S1tb, WtSt2, stb2, SCt, SCtb, 512, 256);

  // 5) Score kernel: row stats, pairwise WMMA GEMMs, median bandwidth, reduce
  rowstats_kernel<<<256, 256, 0, stream>>>(Zi, Zt, SCt, x2v, y2v, ysyv);
  pairwise_kernel<<<16, 256, 0, stream>>>(Zib, SCib, Ztb, SCtb, x2v, y2v, Dist, Gss, Gxsy);
  median_bw_kernel<<<1, 256, 0, stream>>>(Dist, bw2p);
  finalize_kernel<<<1, 256, 0, stream>>>(Dist, Gss, Gxsy, ysyv, KLi, KLt, bw2p,
                                         (float*)d_out);
}
